// QuantizedAttention_53017076301865
// MI455X (gfx1250) — matrix operand order repaired and checked by reading, not compiled, not run on hardware
//
#include <hip/hip_runtime.h>

// Problem constants (match reference)
#define BB 8
#define SS 2048
#define EE 1024
// softmax scale = 1/sqrt(HEAD_DIM=64) = 0.125

typedef __attribute__((ext_vector_type(16))) __bf16 v16bf;
typedef __attribute__((ext_vector_type(8)))  __bf16 v8bf;
typedef __attribute__((ext_vector_type(4)))  __bf16 v4bf;
typedef __attribute__((ext_vector_type(8)))  float  v8f;

union ABfrag { v16bf v; v8bf h[2]; };

// ---------------------------------------------------------------------------
// Per-tensor |max| reduction (uint-bit atomicMax is order-correct for >=0 fp)
// ---------------------------------------------------------------------------
__global__ void __launch_bounds__(256) absmax_kernel(
    const float* __restrict__ w, int n, unsigned* __restrict__ slot)
{
    __shared__ float red[256];
    float m = 0.0f;
    for (int i = blockIdx.x * blockDim.x + threadIdx.x; i < n;
         i += gridDim.x * blockDim.x)
        m = fmaxf(m, fabsf(w[i]));
    red[threadIdx.x] = m;
    __syncthreads();
    for (int s = 128; s > 0; s >>= 1) {
        if (threadIdx.x < s)
            red[threadIdx.x] = fmaxf(red[threadIdx.x], red[threadIdx.x + s]);
        __syncthreads();
    }
    if (threadIdx.x == 0) atomicMax(slot, __float_as_uint(red[0]));
}

// ---------------------------------------------------------------------------
// int8 fake-quant (round-to-nearest-even like jnp.round) -> bf16 weights
// ---------------------------------------------------------------------------
__global__ void __launch_bounds__(256) quant_bf16_kernel(
    const float* __restrict__ w, __bf16* __restrict__ out, int n,
    const unsigned* __restrict__ slot)
{
    const float amax = __uint_as_float(*slot);
    const float s    = amax * (1.0f / 127.0f);
    const float inv  = (s > 0.0f) ? 1.0f / s : 0.0f;
    int i = blockIdx.x * blockDim.x + threadIdx.x;
    if (i < n) {
        float q = rintf(w[i] * inv);
        q = fminf(fmaxf(q, -128.0f), 127.0f);
        out[i] = (__bf16)(q * s);
    }
}

// ---------------------------------------------------------------------------
// fp32 -> bf16, 4-wide
// ---------------------------------------------------------------------------
__global__ void __launch_bounds__(256) f32_to_bf16_x4(
    const float4* __restrict__ in, __bf16* __restrict__ out, int n4)
{
    int i = blockIdx.x * blockDim.x + threadIdx.x;
    if (i < n4) {
        float4 f = in[i];
        v4bf o;
        o[0] = (__bf16)f.x; o[1] = (__bf16)f.y;
        o[2] = (__bf16)f.z; o[3] = (__bf16)f.w;
        *(v4bf*)(out + (size_t)i * 4) = o;
    }
}

// ---------------------------------------------------------------------------
// Generic bf16 WMMA GEMM:  C[m,n] = sum_k A[m,k] * Bm[n,k]   (+ mode extras)
//   A row-major [M,K] stride lda; Bm row-major [N,K] stride ldb (i.e. B^T).
//   One wave -> 32x64 output tile (2 M-subtiles x 4 N-subtiles, 8 acc),
//   software-pipelined with ping-pong fragment buffers so next K-slab loads
//   overlap the 8 WMMAs of the current slab (no full s_wait_loadcnt 0 stalls).
//   MODE 0: store bf16 C[m*ldc+n]
//   MODE 1: store bf16 transposed  CT[n*ldc+m]  (contiguous b128 stores)
//   MODE 2: store f32  C = acc*scale
//   MODE 3: store f32  C = acc + bias[n]
//   gridDim.z batches via element strides (0 for shared operands).
// ---------------------------------------------------------------------------
template <int MODE>
__global__ void __launch_bounds__(256) gemm_bf16_wmma(
    const __bf16* __restrict__ A, const __bf16* __restrict__ Bm,
    void* __restrict__ C, int M, int N, int K, int lda, int ldb, int ldc,
    size_t strideA, size_t strideB, size_t strideC,
    float scale, const float* __restrict__ bias)
{
    const int wave  = blockIdx.x * (blockDim.x >> 5) + (threadIdx.x >> 5);
    const int lane  = threadIdx.x & 31;
    const int ntn   = N >> 6;                  // 64-wide N tiles
    const int tiles = (M >> 5) * ntn;          // 32-tall M tiles
    if (wave >= tiles) return;                  // wave-uniform: EXEC stays full
    const int tm = wave / ntn;
    const int tn = wave % ntn;

    A  += (size_t)blockIdx.z * strideA;
    Bm += (size_t)blockIdx.z * strideB;

    const int r  = lane & 15;                   // A: row, B: column, C: column
    const int hi = lane >> 4;                   // K-half selector

    // A lane: elems 0..7 -> K = k0 + 8*hi .. ; elems 8..15 -> +16
    const __bf16* aPtr0 = A + (size_t)(tm * 32 + r) * lda + hi * 8;
    const __bf16* aPtr1 = aPtr0 + (size_t)16 * lda;
    // B lane: elems 0..7 -> K = k0 + 8*hi .. ; elems 8..15 -> +16
    const __bf16* bPtr  = Bm + (size_t)(tn * 64 + r) * ldb + hi * 8;

    v8f acc[2][4] = {};
    ABfrag a[2][2], b[2][4];

    auto loadA = [&](int buf, int k0) {
        a[buf][0].h[0] = *(const v8bf*)(aPtr0 + k0);
        a[buf][0].h[1] = *(const v8bf*)(aPtr0 + k0 + 16);
        a[buf][1].h[0] = *(const v8bf*)(aPtr1 + k0);
        a[buf][1].h[1] = *(const v8bf*)(aPtr1 + k0 + 16);
    };
    auto loadB = [&](int buf, int k0) {
#pragma unroll
        for (int j = 0; j < 4; ++j) {
            const __bf16* bp = bPtr + (size_t)(j * 16) * ldb + k0;
            b[buf][j].h[0] = *(const v8bf*)(bp);
            b[buf][j].h[1] = *(const v8bf*)(bp + 16);
        }
    };
    auto mma = [&](int buf) {
#pragma unroll
        for (int j = 0; j < 4; ++j)
#pragma unroll
            for (int i = 0; i < 2; ++i)
                acc[i][j] = __builtin_amdgcn_wmma_f32_16x16x32_bf16(
                    false, a[buf][i].v, false, b[buf][j].v, (short)0,
                    acc[i][j], false, false);
    };

    // Software pipeline: K is always a multiple of 64 here (1024 or 2048).
    loadA(0, 0);
    loadB(0, 0);
    for (int k0 = 0; k0 < K; k0 += 64) {
        loadA(1, k0 + 32);
        loadB(1, k0 + 32);
        __builtin_prefetch(aPtr0 + k0 + 64, 0, 1);  // global_prefetch_b8
        __builtin_prefetch(bPtr  + k0 + 64, 0, 1);
        mma(0);                                      // overlaps buf-1 loads
        if (k0 + 64 < K) {
            loadA(0, k0 + 64);
            loadB(0, k0 + 64);
        }
        mma(1);                                      // overlaps buf-0 loads
    }

    // C/D layout: VGPR e, lanes 0-15 -> M=e, lanes 16-31 -> M=e+8, N = lane&15
    if constexpr (MODE == 0) {
        __bf16* Cc = (__bf16*)C + (size_t)blockIdx.z * strideC;
#pragma unroll
        for (int i = 0; i < 2; ++i) {
            const int m_lo = tm * 32 + i * 16 + hi * 8;
#pragma unroll
            for (int j = 0; j < 4; ++j) {
                const int n = tn * 64 + j * 16 + r;
#pragma unroll
                for (int e = 0; e < 8; ++e)
                    Cc[(size_t)(m_lo + e) * ldc + n] = (__bf16)acc[i][j][e];
            }
        }
    } else if constexpr (MODE == 1) {
        __bf16* Cc = (__bf16*)C + (size_t)blockIdx.z * strideC;
#pragma unroll
        for (int i = 0; i < 2; ++i) {
            const int m_lo = tm * 32 + i * 16 + hi * 8;
#pragma unroll
            for (int j = 0; j < 4; ++j) {
                const int n = tn * 64 + j * 16 + r;
                v8bf o;
#pragma unroll
                for (int e = 0; e < 8; ++e) o[e] = (__bf16)acc[i][j][e];
                *(v8bf*)(Cc + (size_t)n * ldc + m_lo) = o;  // 16B-aligned b128
            }
        }
    } else {
        float* Cf = (float*)C + (size_t)blockIdx.z * strideC;
#pragma unroll
        for (int i = 0; i < 2; ++i) {
            const int m_lo = tm * 32 + i * 16 + hi * 8;
#pragma unroll
            for (int j = 0; j < 4; ++j) {
                const int n  = tn * 64 + j * 16 + r;
                const float bv = (MODE == 3) ? bias[n] : 0.0f;
#pragma unroll
                for (int e = 0; e < 8; ++e)
                    Cf[(size_t)(m_lo + e) * ldc + n] = acc[i][j][e] * scale + bv;
            }
        }
    }
}

// ---------------------------------------------------------------------------
// Row softmax over 2048 f32 scores -> bf16 probabilities. One block per row.
// ---------------------------------------------------------------------------
__global__ void __launch_bounds__(256) softmax_rows_2048(
    const float* __restrict__ Sc, __bf16* __restrict__ P)
{
    __shared__ float red[256];
    const size_t base = (size_t)blockIdx.x * 2048;
    const int tid = threadIdx.x;
    float v[8];
    float m = -3.0e38f;
#pragma unroll
    for (int i = 0; i < 8; ++i) {
        v[i] = Sc[base + tid + i * 256];
        m = fmaxf(m, v[i]);
    }
    red[tid] = m; __syncthreads();
    for (int s = 128; s > 0; s >>= 1) {
        if (tid < s) red[tid] = fmaxf(red[tid], red[tid + s]);
        __syncthreads();
    }
    m = red[0]; __syncthreads();
    float sum = 0.0f;
#pragma unroll
    for (int i = 0; i < 8; ++i) { v[i] = __expf(v[i] - m); sum += v[i]; }
    red[tid] = sum; __syncthreads();
    for (int s = 128; s > 0; s >>= 1) {
        if (tid < s) red[tid] += red[tid + s];
        __syncthreads();
    }
    const float inv = 1.0f / red[0];
#pragma unroll
    for (int i = 0; i < 8; ++i)
        P[base + tid + i * 256] = (__bf16)(v[i] * inv);
}

// ---------------------------------------------------------------------------
// Host-side pipeline
// ---------------------------------------------------------------------------
extern "C" void kernel_launch(void* const* d_in, const int* in_sizes, int n_in,
                              void* d_out, int out_size, void* d_ws, size_t ws_size,
                              hipStream_t stream)
{
    (void)in_sizes; (void)n_in; (void)out_size; (void)ws_size;
    const float* x  = (const float*)d_in[0];
    const float* wq = (const float*)d_in[1];
    const float* wk = (const float*)d_in[2];
    const float* wv = (const float*)d_in[3];
    const float* wp = (const float*)d_in[4];
    const float* bp = (const float*)d_in[5];
    float* out = (float*)d_out;

    // Workspace carve-up (256B aligned)
    char* ws = (char*)d_ws;
    size_t off = 0;
    auto carve = [&](size_t bytes) -> void* {
        void* p = ws + off;
        off = (off + bytes + 255) & ~(size_t)255;
        return p;
    };
    unsigned* scaleBits = (unsigned*)carve(4 * sizeof(unsigned));
    __bf16* wq_b = (__bf16*)carve((size_t)EE * EE * 2);
    __bf16* wk_b = (__bf16*)carve((size_t)EE * EE * 2);
    __bf16* wv_b = (__bf16*)carve((size_t)EE * EE * 2);
    __bf16* wp_b = (__bf16*)carve((size_t)EE * EE * 2);
    __bf16* x_b    = (__bf16*)carve((size_t)BB * SS * EE * 2);
    __bf16* q_b    = (__bf16*)carve((size_t)BB * SS * EE * 2);
    __bf16* k_b    = (__bf16*)carve((size_t)BB * SS * EE * 2);
    __bf16* vT_b   = (__bf16*)carve((size_t)BB * EE * SS * 2);  // [b][e][s]
    __bf16* attn_b = (__bf16*)carve((size_t)BB * SS * EE * 2);
    float*  scores = (float*)carve((size_t)BB * SS * SS * 4);
    __bf16* P_b    = (__bf16*)carve((size_t)BB * SS * SS * 2);

    const dim3 blk(256);

    // 1) per-tensor absmax -> scales
    hipMemsetAsync(scaleBits, 0, 4 * sizeof(unsigned), stream);
    const float* wsrc[4] = {wq, wk, wv, wp};
    __bf16*      wdst[4] = {wq_b, wk_b, wv_b, wp_b};
    for (int i = 0; i < 4; ++i)
        absmax_kernel<<<256, blk, 0, stream>>>(wsrc[i], EE * EE, scaleBits + i);
    // 2) int8 fake-quant -> bf16 weights
    for (int i = 0; i < 4; ++i)
        quant_bf16_kernel<<<(EE * EE + 255) / 256, blk, 0, stream>>>(
            wsrc[i], wdst[i], EE * EE, scaleBits + i);
    // 3) x -> bf16
    f32_to_bf16_x4<<<((BB * SS * EE / 4) + 255) / 256, blk, 0, stream>>>(
        (const float4*)x, x_b, BB * SS * EE / 4);

    const int tilesBSE = ((BB * SS) / 32) * (EE / 64);  // 8192
    const int tilesSE  = (SS / 32) * (EE / 64);         // 1024
    const int tilesSSc = (SS / 32) * (SS / 64);         // 2048

    // 4) Q = x * Wq^T ; K = x * Wk^T   (batch folded into M; weights L2-resident)
    gemm_bf16_wmma<0><<<dim3((tilesBSE + 7) / 8, 1, 1), blk, 0, stream>>>(
        x_b, wq_b, q_b, BB * SS, EE, EE, EE, EE, EE, 0, 0, 0, 1.0f, nullptr);
    gemm_bf16_wmma<0><<<dim3((tilesBSE + 7) / 8, 1, 1), blk, 0, stream>>>(
        x_b, wk_b, k_b, BB * SS, EE, EE, EE, EE, EE, 0, 0, 0, 1.0f, nullptr);
    // 5) V^T = (x * Wv^T)^T, batched per b, transposed store -> vT[b][e][s]
    gemm_bf16_wmma<1><<<dim3((tilesSE + 7) / 8, 1, BB), blk, 0, stream>>>(
        x_b, wv_b, vT_b, SS, EE, EE, EE, EE, SS,
        (size_t)SS * EE, 0, (size_t)EE * SS, 1.0f, nullptr);
    // 6) scores = 0.125 * Q K^T  (f32, batched)
    gemm_bf16_wmma<2><<<dim3((tilesSSc + 7) / 8, 1, BB), blk, 0, stream>>>(
        q_b, k_b, scores, SS, SS, EE, EE, EE, SS,
        (size_t)SS * EE, (size_t)SS * EE, (size_t)SS * SS, 0.125f, nullptr);
    // 7) row softmax -> bf16 P
    softmax_rows_2048<<<BB * SS, blk, 0, stream>>>(scores, P_b);
    // 8) attn_out = P * V  via  C[m,n] = sum_k P[m,k] * vT[n,k]
    gemm_bf16_wmma<0><<<dim3((tilesSE + 7) / 8, 1, BB), blk, 0, stream>>>(
        P_b, vT_b, attn_b, SS, EE, SS, SS, SS, EE,
        (size_t)SS * SS, (size_t)EE * SS, (size_t)SS * EE, 1.0f, nullptr);
    // 9) out = attn_out * Wp^T + b_proj  (f32 to d_out)
    gemm_bf16_wmma<3><<<dim3((tilesBSE + 7) / 8, 1, 1), blk, 0, stream>>>(
        attn_b, wp_b, out, BB * SS, EE, EE, EE, EE, EE, 0, 0, 0, 1.0f, bp);
}
